// ChipPairEncoderAlpha_61692910239788
// MI455X (gfx1250) — compile-verified, matrix operand order checked by reading, not run on hardware
//
#include <hip/hip_runtime.h>
#include <hip/hip_bf16.h>

// ---------------------------------------------------------------------------
// ChipPairEncoderAlpha for MI455X (gfx1250, wave32).
// All dense GEMM-shaped work (pair outer-product K=32, conv2 implicit GEMM
// K=576, LN->q/kv projections K=64, axial out-proj K=64) runs on
// v_wmma_f32_16x16x32_f16 (f16 in, f32 accumulate). Softmax / LN / small
// encoder stages stay f32 VALU (negligible FLOPs, latency bound).
// ---------------------------------------------------------------------------

#define B_  4
#define T_  4
#define N_  192
#define CM  32      // C_MSA
#define CP  64      // C_PAIR
#define NH_ 4
#define MROWS (B_ * N_ * N_)   // 147456

typedef __attribute__((ext_vector_type(16))) _Float16 v16h;
typedef __attribute__((ext_vector_type(8)))  float    v8f;

__device__ __forceinline__ float siluf(float x) { return x / (1.f + __expf(-x)); }

// ---------------- track encoder ----------------

// x[b,t,n,c] = sig[b,t,n]*value_w[c] + value_b[c] + track_id[t,c] + pos[n,c]
__global__ void k_init_tracks(const float* __restrict__ sig, const float* __restrict__ vw,
                              const float* __restrict__ vb, const float* __restrict__ tkid,
                              const float* __restrict__ pos, float* __restrict__ x) {
  int idx = blockIdx.x * blockDim.x + threadIdx.x;
  if (idx >= B_ * T_ * N_ * CM) return;
  int c  = idx & 31;
  int n  = (idx >> 5) % N_;
  int bt = idx / (32 * N_);       // b*T + t
  int t  = bt & 3;
  x[idx] = sig[bt * N_ + n] * vw[c] + vb[c] + tkid[t * CM + c] + pos[n * CM + c];
}

// LN over C=32 then qkv = h @ W^T (W is (96,32)); one thread per (row, out-ch)
__global__ void k_ln_qkv32(const float* __restrict__ x, const float* __restrict__ g,
                           const float* __restrict__ bb, const float* __restrict__ w,
                           float* __restrict__ qkv) {
  int idx = blockIdx.x * blockDim.x + threadIdx.x;
  if (idx >= B_ * T_ * N_ * 96) return;
  int co = idx % 96;
  int r  = idx / 96;
  const float* xr = x + r * CM;
  float m = 0.f;
  #pragma unroll
  for (int c = 0; c < CM; ++c) m += xr[c];
  m *= (1.f / CM);
  float v = 0.f;
  #pragma unroll
  for (int c = 0; c < CM; ++c) { float d = xr[c] - m; v += d * d; }
  v *= (1.f / CM);
  float inv = rsqrtf(v + 1e-5f);
  float acc = 0.f;
  #pragma unroll
  for (int c = 0; c < CM; ++c)
    acc += ((xr[c] - m) * inv * g[c] + bb[c]) * w[co * CM + c];
  qkv[idx] = acc;
}

// Spatial attention over N=192 per (b,t), fused out-proj + residual.
__global__ void k_spatial_attn(float* __restrict__ x, const float* __restrict__ qkv,
                               const float* __restrict__ ow) {
  __shared__ float sK[N_ * CM];   // reused for o after barrier
  __shared__ float sV[N_ * CM];
  int bt  = blockIdx.x;
  int tid = threadIdx.x;
  const float* qkvb = qkv + (size_t)bt * N_ * 96;
  for (int i = tid; i < N_ * CM; i += 256) {
    int n = i >> 5, c = i & 31;
    sK[i] = qkvb[n * 96 + 32 + c];
    sV[i] = qkvb[n * 96 + 64 + c];
  }
  __syncthreads();
  float oacc[3][8];
  #pragma unroll
  for (int pi = 0; pi < 3; ++pi) {
    int p = tid + pi * 256;       // 0..767 = (n, head)
    int n = p >> 2, h = (p & 3) * 8;
    float q[8];
    #pragma unroll
    for (int d = 0; d < 8; ++d) q[d] = qkvb[n * 96 + h + d];
    float mx = -3.0e38f, sm = 0.f, acc[8] = {0};
    for (int mi = 0; mi < N_; ++mi) {
      float s = 0.f;
      #pragma unroll
      for (int d = 0; d < 8; ++d) s += q[d] * sK[mi * 32 + h + d];
      s *= 0.35355339059327f;     // 1/sqrt(8)
      if (s > mx) {
        float cr = __expf(mx - s);
        sm *= cr;
        #pragma unroll
        for (int d = 0; d < 8; ++d) acc[d] *= cr;
        mx = s;
      }
      float wt = __expf(s - mx);
      sm += wt;
      #pragma unroll
      for (int d = 0; d < 8; ++d) acc[d] += wt * sV[mi * 32 + h + d];
    }
    float inv = 1.f / sm;
    #pragma unroll
    for (int d = 0; d < 8; ++d) oacc[pi][d] = acc[d] * inv;
  }
  __syncthreads();                // done reading sK/sV
  #pragma unroll
  for (int pi = 0; pi < 3; ++pi) {
    int p = tid + pi * 256;
    int n = p >> 2, h = (p & 3) * 8;
    #pragma unroll
    for (int d = 0; d < 8; ++d) sK[n * 32 + h + d] = oacc[pi][d];
  }
  __syncthreads();
  for (int i = tid; i < N_ * CM; i += 256) {
    int n = i >> 5, c = i & 31;
    float a = 0.f;
    #pragma unroll
    for (int c2 = 0; c2 < CM; ++c2) a += sK[n * 32 + c2] * ow[c * CM + c2];
    x[(size_t)bt * N_ * CM + i] += a;
  }
}

// Track attention over T=4; one thread per (b,n,head) -> writes o (B,T,N,32)
__global__ void k_track_attn(const float* __restrict__ qkv, float* __restrict__ o) {
  int idx = blockIdx.x * blockDim.x + threadIdx.x;
  if (idx >= B_ * N_ * NH_) return;
  int h = (idx & 3) * 8;
  int n = (idx >> 2) % N_;
  int b = idx / (N_ * NH_);
  for (int t = 0; t < T_; ++t) {
    const float* qp = qkv + ((size_t)(b * T_ + t) * N_ + n) * 96 + h;
    float q[8];
    #pragma unroll
    for (int d = 0; d < 8; ++d) q[d] = qp[d];
    float sc[T_];
    #pragma unroll
    for (int s = 0; s < T_; ++s) {
      const float* kp = qkv + ((size_t)(b * T_ + s) * N_ + n) * 96 + 32 + h;
      float a = 0.f;
      #pragma unroll
      for (int d = 0; d < 8; ++d) a += q[d] * kp[d];
      sc[s] = a * 0.35355339059327f;
    }
    float mx = sc[0];
    #pragma unroll
    for (int s = 1; s < T_; ++s) mx = fmaxf(mx, sc[s]);
    float e[T_], sm = 0.f;
    #pragma unroll
    for (int s = 0; s < T_; ++s) { e[s] = __expf(sc[s] - mx); sm += e[s]; }
    float inv = 1.f / sm;
    float* op = o + ((size_t)(b * T_ + t) * N_ + n) * CM + h;
    #pragma unroll
    for (int d = 0; d < 8; ++d) {
      float a = 0.f;
      #pragma unroll
      for (int s = 0; s < T_; ++s)
        a += e[s] * qkv[((size_t)(b * T_ + s) * N_ + n) * 96 + 64 + h + d];
      op[d] = a * inv;
    }
  }
}

// x[r,c] += o[r,:] @ W[c,:]  (W is (32,32))
__global__ void k_proj_res32(float* __restrict__ x, const float* __restrict__ o,
                             const float* __restrict__ w) {
  int idx = blockIdx.x * blockDim.x + threadIdx.x;
  if (idx >= B_ * T_ * N_ * CM) return;
  int c = idx & 31;
  int r = idx >> 5;
  const float* orow = o + (size_t)r * CM;
  float acc = 0.f;
  #pragma unroll
  for (int c2 = 0; c2 < CM; ++c2) acc += orow[c2] * w[c * CM + c2];
  x[idx] += acc;
}

// softmax gate over T, pooled output (B,N,32); one thread per (b,n)
__global__ void k_gate(const float* __restrict__ x, const float* __restrict__ gw,
                       const float* __restrict__ gb, float* __restrict__ out) {
  int idx = blockIdx.x * blockDim.x + threadIdx.x;
  if (idx >= B_ * N_) return;
  int n = idx % N_, b = idx / N_;
  float lg[T_];
  #pragma unroll
  for (int t = 0; t < T_; ++t) {
    const float* xr = x + ((size_t)(b * T_ + t) * N_ + n) * CM;
    float a = gb[0];
    #pragma unroll
    for (int c = 0; c < CM; ++c) a += xr[c] * gw[c];
    lg[t] = a;
  }
  float mx = lg[0];
  #pragma unroll
  for (int t = 1; t < T_; ++t) mx = fmaxf(mx, lg[t]);
  float e[T_], sm = 0.f;
  #pragma unroll
  for (int t = 0; t < T_; ++t) { e[t] = __expf(lg[t] - mx); sm += e[t]; }
  float inv = 1.f / sm;
  for (int c = 0; c < CM; ++c) {
    float a = 0.f;
    #pragma unroll
    for (int t = 0; t < T_; ++t)
      a += e[t] * x[((size_t)(b * T_ + t) * N_ + n) * CM + c];
    out[((size_t)b * N_ + n) * CM + c] = a * inv;
  }
}

// ---------------- pair outer product (WMMA) ----------------

// t_h[b, i*64+p, d] = sum_c row_r[b,i,c] * pair_w[p,c,d]   (f16 out)
__global__ void k_pair_t(const float* __restrict__ rr, const float* __restrict__ pw,
                         _Float16* __restrict__ th) {
  int idx = blockIdx.x * blockDim.x + threadIdx.x;
  if (idx >= B_ * N_ * 64 * 32) return;
  int dd = idx & 31;
  int p  = (idx >> 5) & 63;
  int i  = (idx >> 11) % N_;
  int b  = idx / (N_ * 64 * 32);
  const float* r = rr + ((size_t)b * N_ + i) * CM;
  float acc = 0.f;
  #pragma unroll
  for (int c = 0; c < CM; ++c) acc += r[c] * pw[p * 1024 + c * 32 + dd];
  th[idx] = (_Float16)acc;
}

// colB_h[b, d, j] = col_r[b, j, d]   (f16 transposed B-matrix: K x Ncols)
__global__ void k_colB(const float* __restrict__ cr, _Float16* __restrict__ colB) {
  int idx = blockIdx.x * blockDim.x + threadIdx.x;
  if (idx >= B_ * 32 * N_) return;
  int j  = idx % N_;
  int dd = (idx / N_) & 31;
  int b  = idx / (32 * N_);
  colB[idx] = (_Float16)cr[((size_t)b * N_ + j) * CM + dd];
}

// pair[b,i,j,p] = sum_d t_h[b,(i,p),d] * colB_h[b,d,j] + bias[p]
// One 16x16 tile per wave; K=32 -> single v_wmma_f32_16x16x32_f16.
__global__ void k_pair_wmma(const _Float16* __restrict__ A, const _Float16* __restrict__ Bm,
                            const float* __restrict__ bias, float* __restrict__ pair) {
  int lane = threadIdx.x & 31;
  int wave = threadIdx.x >> 5;
  int tile = blockIdx.x * 8 + wave;                 // 0..36863
  int b    = tile / (768 * 12);
  int rem  = tile % (768 * 12);
  int mt   = rem / 12, nt = rem % 12;
  int rowbase = mt * 16, colbase = nt * 16;
  const _Float16* Ab = A  + (size_t)b * 12288 * 32;
  const _Float16* Bb = Bm + (size_t)b * 32 * N_;
  int lm = lane & 15;
  int hk = (lane >> 4) * 8;                         // lane-group K offset
  v16h a, bf;
  // CDNA5 16-bit A(16x32)/B(32x16) fragment layout:
  // element e -> k = (e>=8 ? 16 : 0) + (lane>=16 ? 8 : 0) + (e & 7)
  #pragma unroll
  for (int e = 0; e < 16; ++e) {
    int k = ((e >> 3) << 4) + hk + (e & 7);
    a[e]  = Ab[(size_t)(rowbase + lm) * 32 + k];
    bf[e] = Bb[(size_t)k * N_ + colbase + lm];
  }
  v8f c = {};
  v8f dacc = __builtin_amdgcn_wmma_f32_16x16x32_f16(
      false, a, false, bf, (short)0, c, false, false);
  #pragma unroll
  for (int r = 0; r < 8; ++r) {
    int m = r + ((lane >> 4) << 3);
    int R = rowbase + m;                            // R = i*64 + p
    int p = R & 63, i = R >> 6;
    int j = colbase + lm;
    pair[(((size_t)b * N_ + i) * N_ + j) * CP + p] = dacc[r] + bias[p];
  }
}

// ---------------- generic f16 WMMA GEMM ----------------
// out[M x Nc] (f32) = A_h[M x Ktot] @ Bm_h[Ktot x Nc]; K multiple of 32.
// addTo: accumulate into out.  transposeOut: remap row (b,a,w)->(b,w,a).
__global__ void k_gemm_wmma(const _Float16* __restrict__ A, const _Float16* __restrict__ Bm,
                            float* __restrict__ out, int Nc, int Ktot,
                            int addTo, int transposeOut) {
  int lane = threadIdx.x & 31;
  int wave = threadIdx.x >> 5;
  int tile = blockIdx.x * 8 + wave;
  int ntn  = Nc >> 4;
  int mt   = tile / ntn, nt = tile % ntn;
  int rowbase = mt * 16, colbase = nt * 16;
  int lm = lane & 15;
  int hk = (lane >> 4) * 8;
  v8f c = {};
  for (int kk = 0; kk < Ktot; kk += 32) {
    v16h a, bf;
    #pragma unroll
    for (int e = 0; e < 16; ++e) {
      int k = kk + ((e >> 3) << 4) + hk + (e & 7);
      a[e]  = A[(size_t)(rowbase + lm) * Ktot + k];
      bf[e] = Bm[(size_t)k * Nc + colbase + lm];
    }
    c = __builtin_amdgcn_wmma_f32_16x16x32_f16(false, a, false, bf, (short)0, c, false, false);
  }
  #pragma unroll
  for (int r = 0; r < 8; ++r) {
    int m = r + ((lane >> 4) << 3);
    int R = rowbase + m;
    if (transposeOut) {
      int b = R / (N_ * N_);
      int rem = R % (N_ * N_);
      R = b * (N_ * N_) + (rem % N_) * N_ + (rem / N_);
    }
    size_t oi = (size_t)R * Nc + colbase + lm;
    if (addTo) out[oi] += c[r]; else out[oi] = c[r];
  }
}

// ---------------- conv stack ----------------

// h1[b,oc,y,x] = silu(conv3x3(bulk)[..] + b1[oc])
__global__ void k_conv1(const float* __restrict__ bulk, const float* __restrict__ w1,
                        const float* __restrict__ b1, float* __restrict__ h1) {
  int idx = blockIdx.x * blockDim.x + threadIdx.x;
  if (idx >= B_ * CP * N_ * N_) return;
  int x  = idx % N_;
  int y  = (idx / N_) % N_;
  int oc = (idx / (N_ * N_)) % CP;
  int b  = idx / (N_ * N_ * CP);
  float acc = b1[oc];
  const float* bp = bulk + (size_t)b * N_ * N_;
  #pragma unroll
  for (int ky = 0; ky < 3; ++ky) {
    int yy = y + ky - 1;
    if (yy < 0 || yy >= N_) continue;
    #pragma unroll
    for (int kx = 0; kx < 3; ++kx) {
      int xx = x + kx - 1;
      if (xx < 0 || xx >= N_) continue;
      acc += bp[yy * N_ + xx] * w1[oc * 9 + ky * 3 + kx];
    }
  }
  h1[idx] = siluf(acc);
}

// conv2 weights -> f16 B matrix: Bw[k=ic*9+r][oc] = w2[oc,ic,r]
__global__ void k_w16conv(const float* __restrict__ w2, _Float16* __restrict__ bw) {
  int idx = blockIdx.x * blockDim.x + threadIdx.x;
  if (idx >= 576 * CP) return;
  int oc = idx & 63;
  int k  = idx >> 6;
  int ic = k / 9;
  int rr = k - ic * 9;
  bw[idx] = (_Float16)w2[(oc * CP + ic) * 9 + rr];
}

// ctx[b,y,x,oc] = im2col(h1) @ Bw + b2[oc]  -- implicit GEMM, K=576 (18 WMMAs)
__global__ void k_conv2_wmma(const float* __restrict__ h1, const _Float16* __restrict__ Bw,
                             const float* __restrict__ b2, float* __restrict__ ctx) {
  int lane = threadIdx.x & 31;
  int wave = threadIdx.x >> 5;
  int tile = blockIdx.x * 8 + wave;   // (MROWS/16)*(CP/16) = 36864 tiles
  int mt = tile >> 2, nt = tile & 3;
  int rowbase = mt * 16, ocbase = nt * 16;
  int lm = lane & 15;
  int hk = (lane >> 4) * 8;
  int R = rowbase + lm;               // this lane's A row = (b,y,x)
  int x = R % N_;
  int y = (R / N_) % N_;
  int b = R / (N_ * N_);
  v8f c = {};
  for (int kk = 0; kk < 576; kk += 32) {
    v16h a, bf;
    #pragma unroll
    for (int e = 0; e < 16; ++e) {
      int k  = kk + ((e >> 3) << 4) + hk + (e & 7);
      int ic = k / 9;
      int rr = k - ic * 9;
      int ky = rr / 3, kx = rr - ky * 3;
      int yy = y + ky - 1, xx = x + kx - 1;
      float v = 0.f;
      if (yy >= 0 && yy < N_ && xx >= 0 && xx < N_)
        v = h1[((size_t)(b * CP + ic) * N_ + yy) * N_ + xx];
      a[e]  = (_Float16)v;
      bf[e] = Bw[(size_t)k * CP + ocbase + lm];
    }
    c = __builtin_amdgcn_wmma_f32_16x16x32_f16(false, a, false, bf, (short)0, c, false, false);
  }
  #pragma unroll
  for (int r = 0; r < 8; ++r) {
    int m  = r + ((lane >> 4) << 3);
    int Rw = rowbase + m;
    int oc = ocbase + lm;
    ctx[(size_t)Rw * CP + oc] = c[r] + b2[oc];
  }
}

// ---------------- axial cross attention ----------------

// LN over C=64 -> f16 activations (GEMM A-matrix)
__global__ void k_ln16(const float* __restrict__ in, const float* __restrict__ g,
                       const float* __restrict__ bb, _Float16* __restrict__ out) {
  int r = blockIdx.x * blockDim.x + threadIdx.x;
  if (r >= MROWS) return;
  const float* xr = in + (size_t)r * CP;
  float m = 0.f;
  #pragma unroll
  for (int c = 0; c < CP; ++c) m += xr[c];
  m *= (1.f / CP);
  float v = 0.f;
  #pragma unroll
  for (int c = 0; c < CP; ++c) { float d = xr[c] - m; v += d * d; }
  v *= (1.f / CP);
  float inv = rsqrtf(v + 1e-5f);
  _Float16* op = out + (size_t)r * CP;
  #pragma unroll
  for (int c = 0; c < CP; ++c)
    op[c] = (_Float16)((xr[c] - m) * inv * g[c] + bb[c]);
}

// weight (outC,64) -> f16 B matrix (64 x outC): Bm[k][n] = W[n][k]
__global__ void k_w16b(const float* __restrict__ w, _Float16* __restrict__ bm, int outC) {
  int idx = blockIdx.x * blockDim.x + threadIdx.x;
  if (idx >= CP * outC) return;
  int n = idx % outC;
  int k = idx / outC;
  bm[idx] = (_Float16)w[n * CP + k];
}

// One block per (b, axis-index a); pairs (w, head); online softmax over v.
// transposed=0: attend along x within row a; =1: attend along y within col a.
// Output o is f16 (feeds the out-proj WMMA GEMM).
__global__ void k_axial_attn(const float* __restrict__ q, const float* __restrict__ kv,
                             const float* __restrict__ rel, _Float16* __restrict__ o,
                             int transposed) {
  int b   = blockIdx.x / N_;
  int a   = blockIdx.x % N_;
  int tid = threadIdx.x;
  #pragma unroll
  for (int pi = 0; pi < 3; ++pi) {
    int p = tid + pi * 256;       // 0..767 = (w, head)
    int w = p >> 2;
    int head = p & 3;
    int h = head * 16;
    int yq = transposed ? w : a, xq = transposed ? a : w;
    const float* qp = q + (((size_t)b * N_ + yq) * N_ + xq) * CP + h;
    float qv[16];
    #pragma unroll
    for (int d = 0; d < 16; ++d) qv[d] = qp[d];
    const float* relp = rel + (191 - w) * NH_ + head;
    float mx = -3.0e38f, sm = 0.f, acc[16] = {0};
    for (int vv = 0; vv < N_; ++vv) {
      int yk = transposed ? vv : a, xk = transposed ? a : vv;
      const float* kp = kv + (((size_t)b * N_ + yk) * N_ + xk) * 128 + h;
      float s = 0.f;
      #pragma unroll
      for (int d = 0; d < 16; ++d) s += qv[d] * kp[d];
      s = s * 0.25f + relp[vv * NH_];     // 1/sqrt(16) + rel[v-w+191, head]
      if (s > mx) {
        float cr = __expf(mx - s);
        sm *= cr;
        #pragma unroll
        for (int d = 0; d < 16; ++d) acc[d] *= cr;
        mx = s;
      }
      float wt = __expf(s - mx);
      sm += wt;
      const float* vp = kp + 64;          // v half of kv
      #pragma unroll
      for (int d = 0; d < 16; ++d) acc[d] += wt * vp[d];
    }
    float inv = 1.f / sm;
    _Float16* op = o + (((size_t)b * N_ + a) * N_ + w) * CP + h;
    #pragma unroll
    for (int d = 0; d < 16; ++d) op[d] = (_Float16)(acc[d] * inv);
  }
}

// pair += alpha*LN(pair); silu; transpose to (B, 64, N, N)
__global__ void k_final(const float* __restrict__ pair, const float* __restrict__ g,
                        const float* __restrict__ bb, const float* __restrict__ alpha,
                        float* __restrict__ out) {
  int r = blockIdx.x * blockDim.x + threadIdx.x;
  if (r >= MROWS) return;
  const float* pr = pair + (size_t)r * CP;
  float m = 0.f;
  #pragma unroll
  for (int c = 0; c < CP; ++c) m += pr[c];
  m *= (1.f / CP);
  float v = 0.f;
  #pragma unroll
  for (int c = 0; c < CP; ++c) { float d = pr[c] - m; v += d * d; }
  v *= (1.f / CP);
  float inv = rsqrtf(v + 1e-5f);
  float al = alpha[0];
  int x = r % N_, y = (r / N_) % N_, b = r / (N_ * N_);
  for (int p = 0; p < CP; ++p) {
    float u = pr[p] + al * ((pr[p] - m) * inv * g[p] + bb[p]);
    out[(((size_t)b * CP + p) * N_ + y) * N_ + x] = siluf(u);
  }
}

// ---------------------------------------------------------------------------

extern "C" void kernel_launch(void* const* d_in, const int* in_sizes, int n_in,
                              void* d_out, int out_size, void* d_ws, size_t ws_size,
                              hipStream_t stream) {
  (void)in_sizes; (void)n_in; (void)out_size; (void)ws_size;
  // --- inputs (setup_inputs insertion order, recursively flattened) ---
  const float* row_tracks = (const float*)d_in[0];
  const float* col_tracks = (const float*)d_in[1];
  const float* bulk       = (const float*)d_in[2];
  const float* track_id   = (const float*)d_in[3];
  const float* value_w    = (const float*)d_in[4];
  const float* value_b    = (const float*)d_in[5];
  const float* pos_row    = (const float*)d_in[6];
  const float* pos_col    = (const float*)d_in[7];
  // blocks occupy indices 8..23 (2 blocks x {spatial,track} x {ng,nb,qkv_w,out_w})
  const float* gate_w     = (const float*)d_in[24];
  const float* gate_b     = (const float*)d_in[25];
  const float* pair_w     = (const float*)d_in[26];
  const float* pair_b     = (const float*)d_in[27];
  const float* conv1_w    = (const float*)d_in[28];
  const float* conv1_b    = (const float*)d_in[29];
  const float* conv2_w    = (const float*)d_in[30];
  const float* conv2_b    = (const float*)d_in[31];
  // xrow 32..39, xcol 40..47 : {qg,qb,kg,kb,wq,wkv,wo,rel}
  const float* ada_g      = (const float*)d_in[48];
  const float* ada_b      = (const float*)d_in[49];
  const float* ada_alpha  = (const float*)d_in[50];
  float* out = (float*)d_out;
  float* ws  = (float*)d_ws;

  // --- workspace carve-up (floats) ---
  size_t off = 0;
  float* xbuf   = ws + off; off += (size_t)B_ * T_ * N_ * CM;
  float* qkvbuf = ws + off; off += (size_t)B_ * T_ * N_ * 96;
  float* obuf_e = ws + off; off += (size_t)B_ * T_ * N_ * CM;
  float* row_r  = ws + off; off += (size_t)B_ * N_ * CM;
  float* col_r  = ws + off; off += (size_t)B_ * N_ * CM;
  _Float16* t_h    = (_Float16*)(ws + off); off += ((size_t)B_ * N_ * 64 * 32) / 2;
  _Float16* colB_h = (_Float16*)(ws + off); off += ((size_t)B_ * 32 * N_) / 2;
  _Float16* wsm_h  = (_Float16*)(ws + off); off += ((size_t)CP * 128) / 2;     // wq/wkv/wo f16
  _Float16* wcv_h  = (_Float16*)(ws + off); off += ((size_t)576 * CP) / 2;     // conv2 w f16
  _Float16* hn_h   = (_Float16*)(ws + off); off += ((size_t)MROWS * CP) / 2;   // LN'd acts f16
  float* pairb  = ws + off; off += (size_t)MROWS * CP;        // 9.44M
  float* ctx    = ws + off; off += (size_t)MROWS * CP;        // 9.44M
  float* qbuf   = ws + off; off += (size_t)MROWS * CP;        // 9.44M
  float* kvbuf  = ws + off; off += (size_t)MROWS * 128;       // 18.9M
  float* h1     = ws + off;                                   // aliased:
  _Float16* o_h = (_Float16*)h1; off += (size_t)MROWS * CP;   // h1 dead before o_h use

  // --- track encoders (row then col) ---
  const float* sigs[2]  = {row_tracks, col_tracks};
  const float* poss[2]  = {pos_row, pos_col};
  float*       dests[2] = {row_r, col_r};
  const int nX = B_ * T_ * N_ * CM;
  const int nQ = B_ * T_ * N_ * 96;
  for (int s = 0; s < 2; ++s) {
    k_init_tracks<<<(nX + 255) / 256, 256, 0, stream>>>(sigs[s], value_w, value_b,
                                                        track_id, poss[s], xbuf);
    for (int blk = 0; blk < 2; ++blk) {
      int base = 8 + blk * 8;
      const float* sp_ng  = (const float*)d_in[base + 0];
      const float* sp_nb  = (const float*)d_in[base + 1];
      const float* sp_qkv = (const float*)d_in[base + 2];
      const float* sp_ow  = (const float*)d_in[base + 3];
      const float* tr_ng  = (const float*)d_in[base + 4];
      const float* tr_nb  = (const float*)d_in[base + 5];
      const float* tr_qkv = (const float*)d_in[base + 6];
      const float* tr_ow  = (const float*)d_in[base + 7];
      k_ln_qkv32<<<(nQ + 255) / 256, 256, 0, stream>>>(xbuf, sp_ng, sp_nb, sp_qkv, qkvbuf);
      k_spatial_attn<<<B_ * T_, 256, 0, stream>>>(xbuf, qkvbuf, sp_ow);
      k_ln_qkv32<<<(nQ + 255) / 256, 256, 0, stream>>>(xbuf, tr_ng, tr_nb, tr_qkv, qkvbuf);
      k_track_attn<<<(B_ * N_ * NH_ + 255) / 256, 256, 0, stream>>>(qkvbuf, obuf_e);
      k_proj_res32<<<(nX + 255) / 256, 256, 0, stream>>>(xbuf, obuf_e, tr_ow);
    }
    k_gate<<<(B_ * N_ + 255) / 256, 256, 0, stream>>>(xbuf, gate_w, gate_b, dests[s]);
  }

  // --- pair outer product: t = row_r @ pair_w ; pair = t @ col_r^T (WMMA) ---
  { int n = B_ * N_ * 64 * 32;
    k_pair_t<<<(n + 255) / 256, 256, 0, stream>>>(row_r, pair_w, t_h); }
  { int n = B_ * 32 * N_;
    k_colB<<<(n + 255) / 256, 256, 0, stream>>>(col_r, colB_h); }
  k_pair_wmma<<<4608, 256, 0, stream>>>(t_h, colB_h, pair_b, pairb);

  // --- conv stack: conv1 (VALU, tiny K) then conv2 as implicit-GEMM WMMA ---
  { int n = B_ * CP * N_ * N_;
    k_conv1<<<(n + 255) / 256, 256, 0, stream>>>(bulk, conv1_w, conv1_b, h1); }
  k_w16conv<<<(576 * CP + 255) / 256, 256, 0, stream>>>(conv2_w, wcv_h);
  k_conv2_wmma<<<4608, 256, 0, stream>>>(h1, wcv_h, conv2_b, ctx);

  // --- axial cross attention: row (ax=0) then col (ax=1) ---
  const int M = MROWS;
  for (int ax = 0; ax < 2; ++ax) {
    int base = 32 + ax * 8;
    const float* qg  = (const float*)d_in[base + 0];
    const float* qb  = (const float*)d_in[base + 1];
    const float* kg  = (const float*)d_in[base + 2];
    const float* kb  = (const float*)d_in[base + 3];
    const float* wq  = (const float*)d_in[base + 4];
    const float* wkv = (const float*)d_in[base + 5];
    const float* wo  = (const float*)d_in[base + 6];
    const float* rel = (const float*)d_in[base + 7];
    // q = LN(pair) @ wq^T   (WMMA GEMM, K=64, Nc=64)
    k_w16b<<<(CP * 64 + 255) / 256, 256, 0, stream>>>(wq, wsm_h, 64);
    k_ln16<<<(M + 255) / 256, 256, 0, stream>>>(pairb, qg, qb, hn_h);
    k_gemm_wmma<<<(M / 16) * 4 / 8, 256, 0, stream>>>(hn_h, wsm_h, qbuf, 64, 64, 0, 0);
    // kv = LN(ctx) @ wkv^T  (WMMA GEMM, K=64, Nc=128)
    k_w16b<<<(CP * 128 + 255) / 256, 256, 0, stream>>>(wkv, wsm_h, 128);
    k_ln16<<<(M + 255) / 256, 256, 0, stream>>>(ctx, kg, kb, hn_h);
    k_gemm_wmma<<<(M / 16) * 8 / 8, 256, 0, stream>>>(hn_h, wsm_h, kvbuf, 128, 64, 0, 0);
    // attention (online softmax, f16 o out)
    k_axial_attn<<<B_ * N_, 256, 0, stream>>>(qbuf, kvbuf, rel, o_h, ax);
    // pair += o @ wo^T      (WMMA GEMM with residual accumulate, transpose on col pass)
    k_w16b<<<(CP * 64 + 255) / 256, 256, 0, stream>>>(wo, wsm_h, 64);
    k_gemm_wmma<<<(M / 16) * 4 / 8, 256, 0, stream>>>(o_h, wsm_h, pairb, 64, 64, 1, ax);
  }

  // --- adaLN + silu + NCHW transpose ---
  k_final<<<(M + 255) / 256, 256, 0, stream>>>(pairb, ada_g, ada_b, ada_alpha, out);
}
